// FuzzyMultiheadAttention_69200513073607
// MI455X (gfx1250) — matrix operand order repaired and checked by reading, not compiled, not run on hardware
//
#include <hip/hip_runtime.h>
#include <hip/hip_bf16.h>

typedef __attribute__((ext_vector_type(4)))  float  v4f;
typedef __attribute__((ext_vector_type(8)))  float  v8f;
typedef __attribute__((ext_vector_type(4)))  __bf16 v4bf;
typedef __attribute__((ext_vector_type(8)))  __bf16 v8bf;
typedef __attribute__((ext_vector_type(16))) __bf16 v16bf;

#define WMMA_BF16(a, b, c) \
  __builtin_amdgcn_wmma_f32_16x16x32_bf16(false, (a), false, (b), (short)0, (c), false, false)

// ---------------------------------------------------------------------------
// bf16x3 split-precision WMMA GEMM core (fp32-faithful on the fast path):
//   C[128x128] += A[m0:+128, :K] * W[n0:+128, :K]^T
//   A = Ah + Al (bf16 hi/lo), W = Wh + Wl;  C ~= Ah*Wh + Ah*Wl + Al*Wh (fp32 acc)
// 256 threads / 8 waves. Wave w: rows 32*(w%4), cols 64*(w/4) -> 2x4 16x16 tiles.
// LDS row stride 40 bf16 (80B): 16B-aligned b128 fragment reads, conflict-free.
// ---------------------------------------------------------------------------
#define BK   32
#define LSTR 40

__device__ __forceinline__ v16bf fragA(const __bf16* rowp, int lh) {
  // ISA 16-bit A 16x32: lanes 0-15 hold K[0..7],K[16..23]; lanes 16-31 K[8..15],K[24..31]
  v8bf c0 = *(const v8bf*)(rowp + 8 * lh);
  v8bf c1 = *(const v8bf*)(rowp + 16 + 8 * lh);
  return __builtin_shufflevector(c0, c1, 0,1,2,3,4,5,6,7,8,9,10,11,12,13,14,15);
}
__device__ __forceinline__ v16bf fragB(const __bf16* colp, int lh) {
  // ISA 16-bit B 32x16: lanes 0-15 hold K[0..7],K[16..23]; lanes 16-31 K[8..15],K[24..31]
  v8bf c0 = *(const v8bf*)(colp + 8 * lh);
  v8bf c1 = *(const v8bf*)(colp + 16 + 8 * lh);
  return __builtin_shufflevector(c0, c1, 0,1,2,3,4,5,6,7,8,9,10,11,12,13,14,15);
}

__device__ __forceinline__ void split_store(v4f v, __bf16* hi, __bf16* lo) {
  v4bf h, l;
#pragma unroll
  for (int j = 0; j < 4; ++j) {
    float x  = v[j];
    __bf16 hb = (__bf16)x;            // RNE
    h[j] = hb;
    l[j] = (__bf16)(x - (float)hb);   // residual
  }
  *(v4bf*)hi = h;
  *(v4bf*)lo = l;
}

__device__ __forceinline__ void gemm_core_bf16x3(
    const float* __restrict__ A, const float* __restrict__ W, int K,
    int m0, int n0,
    __bf16* Ah, __bf16* Al, __bf16* Bh, __bf16* Bl,
    v8f (&acc)[2][4])
{
  const int tid   = threadIdx.x;
  const int lane  = tid & 31;
  const int w     = tid >> 5;
  const int lh    = lane >> 4;
  const int lidx  = lane & 15;
  const int rbase = (w & 3) << 5;   // 0,32,64,96
  const int cbase = (w >> 2) << 6;  // 0,64

#pragma unroll
  for (int i = 0; i < 2; ++i)
#pragma unroll
    for (int j = 0; j < 4; ++j)
      acc[i][j] = (v8f){0.f,0.f,0.f,0.f,0.f,0.f,0.f,0.f};

  for (int k0 = 0; k0 < K; k0 += BK) {
#pragma unroll
    for (int i = 0; i < 4; ++i) {         // 1024 float4s per 128x32 fp32 tile
      int linear = tid + i * 256;
      int r  = linear >> 3;
      int c4 = (linear & 7) << 2;
      v4f av = *(const v4f*)(A + (size_t)(m0 + r) * K + k0 + c4);
      split_store(av, Ah + r * LSTR + c4, Al + r * LSTR + c4);
      v4f bv = *(const v4f*)(W + (size_t)(n0 + r) * K + k0 + c4);
      split_store(bv, Bh + r * LSTR + c4, Bl + r * LSTR + c4);
    }
    __syncthreads();

    v16bf ah[2], al[2];
#pragma unroll
    for (int i = 0; i < 2; ++i) {
      int roff = (rbase + 16 * i + lidx) * LSTR;
      ah[i] = fragA(Ah + roff, lh);
      al[i] = fragA(Al + roff, lh);
    }
#pragma unroll
    for (int j = 0; j < 4; ++j) {
      int coff = (cbase + 16 * j + lidx) * LSTR;
      v16bf bh = fragB(Bh + coff, lh);
      v16bf bl = fragB(Bl + coff, lh);
#pragma unroll
      for (int i = 0; i < 2; ++i) {
        acc[i][j] = WMMA_BF16(ah[i], bh, acc[i][j]);
        acc[i][j] = WMMA_BF16(ah[i], bl, acc[i][j]);
        acc[i][j] = WMMA_BF16(al[i], bh, acc[i][j]);
      }
    }
    __syncthreads();
  }
}

#define DECL_GEMM_LDS                                   \
  __shared__ __align__(16) __bf16 Ah[128 * LSTR];       \
  __shared__ __align__(16) __bf16 Al[128 * LSTR];       \
  __shared__ __align__(16) __bf16 Bh[128 * LSTR];       \
  __shared__ __align__(16) __bf16 Bl[128 * LSTR];

// ---------------------------------------------------------------------------
// 1) Q projection: qp = (query @ Wq^T + bq) * D^-0.5        (2048x1024)
// ---------------------------------------------------------------------------
__global__ void __launch_bounds__(256) k_qproj(const float* __restrict__ query,
                                               const float* __restrict__ Wq,
                                               const float* __restrict__ bq,
                                               float* __restrict__ qp)
{
  DECL_GEMM_LDS
  const int m0 = blockIdx.y * 128;
  const int n0 = blockIdx.x * 128;
  v8f acc[2][4];
  gemm_core_bf16x3(query, Wq, 1024, m0, n0, Ah, Al, Bh, Bl, acc);

  const int lane = threadIdx.x & 31;
  const int w    = threadIdx.x >> 5;
  const int lh   = lane >> 4;
  const int lidx = lane & 15;
  const int rbase = (w & 3) << 5;
  const int cbase = (w >> 2) << 6;
  const float scale = 0.125f;   // 64^-0.5
#pragma unroll
  for (int j = 0; j < 4; ++j) {
    int n = n0 + cbase + 16 * j + lidx;
    float bj = bq[n];
#pragma unroll
    for (int i = 0; i < 2; ++i)
#pragma unroll
      for (int e = 0; e < 8; ++e) {
        int row = m0 + rbase + 16 * i + e + 8 * lh;
        qp[(size_t)row * 1024 + n] = (acc[i][j][e] + bj) * scale;
      }
  }
}

// ---------------------------------------------------------------------------
// 2) Fuzzy Gaussian membership + softmax over rules.
//    One wave per (token, head); lane = rule index r (wave32 == R).
// ---------------------------------------------------------------------------
__global__ void __launch_bounds__(256) k_fuzzy(const float* __restrict__ qp,
                                               const float* __restrict__ rk,
                                               const float* __restrict__ rw,
                                               float* __restrict__ attn)
{
  const int w    = threadIdx.x >> 5;
  const int lane = threadIdx.x & 31;
  const int item = blockIdx.x * 8 + w;   // 0 .. 2048*16-1
  const int t = item >> 4;
  const int h = item & 15;

  const float* qv  = qp + (size_t)t * 1024 + h * 64;
  const float* rkp = rk + ((size_t)h * 32 + lane) * 64;
  const float* rwp = rw + ((size_t)h * 32 + lane) * 64;

  float z = 0.f;
#pragma unroll 8
  for (int d = 0; d < 64; ++d) {
    float diff = fabsf(qv[d] - rkp[d]) / rwp[d];
    z = fmaf(diff, diff, z);
  }
  z *= -0.5f / 64.f;   // HTSK mean over head_dim, then -0.5

  float m = z;
#pragma unroll
  for (int x = 1; x < 32; x <<= 1) m = fmaxf(m, __shfl_xor(m, x, 32));
  float e = __expf(z - m);
  float s = e;
#pragma unroll
  for (int x = 1; x < 32; x <<= 1) s += __shfl_xor(s, x, 32);
  attn[(size_t)item * 32 + lane] = e / s;
}

// ---------------------------------------------------------------------------
// 3) V projection fused with rule-weighted aggregation.
//    vproj[t, n], n = h*D*R + d*R + r. Each wave's 16-col tile pair j={0,1}
//    (and j={2,3}) covers one full r-group = one (h,d) pair, so accumulators
//    are weighted by LDS-staged attn and reduced across 16 lanes; the 268 MB
//    vproj intermediate never touches memory. Output stored head-major
//    (b,h,s,d) = the reference's faithful transpose(1,2).view flatten.
// ---------------------------------------------------------------------------
__global__ void __launch_bounds__(256) k_vproj(const float* __restrict__ value,
                                               const float* __restrict__ Wv,
                                               const float* __restrict__ bv,
                                               const float* __restrict__ attn,
                                               float* __restrict__ X)
{
  DECL_GEMM_LDS
  __shared__ __align__(16) float ats[128 * 32];   // attn tile: 128 tokens x 32 rules
  const int m0 = blockIdx.y * 128;  // token tile
  const int n0 = blockIdx.x * 128;  // vproj column tile (within one head)
  const int h  = n0 >> 11;          // n0 / (D*R)

  for (int i = threadIdx.x; i < 1024; i += 256) {  // 128x32 floats as float4
    int t  = i >> 3;
    int c4 = (i & 7) << 2;
    *(v4f*)(ats + t * 32 + c4) =
        *(const v4f*)(attn + ((size_t)(m0 + t) * 16 + h) * 32 + c4);
  }
  // (gemm_core's first __syncthreads covers the ats hazard)

  v8f acc[2][4];
  gemm_core_bf16x3(value, Wv, 1024, m0, n0, Ah, Al, Bh, Bl, acc);

  const int lane = threadIdx.x & 31;
  const int w    = threadIdx.x >> 5;
  const int lh   = lane >> 4;
  const int lidx = lane & 15;
  const int rbase = (w & 3) << 5;
  const int cbase = (w >> 2) << 6;
  const int d0 = ((n0 + cbase) >> 5) & 63;  // first (h,d) pair of this wave
  float bvj[4];
#pragma unroll
  for (int j = 0; j < 4; ++j) bvj[j] = bv[n0 + cbase + 16 * j + lidx];

#pragma unroll
  for (int i = 0; i < 2; ++i)
#pragma unroll
    for (int e = 0; e < 8; ++e) {
      int row = rbase + 16 * i + e + 8 * lh;          // local token 0..127
      float a0 = ats[row * 32 + lidx];                // r = lidx
      float a1 = ats[row * 32 + 16 + lidx];           // r = 16+lidx
      float t0 = a0 * (acc[i][0][e] + bvj[0]) + a1 * (acc[i][1][e] + bvj[1]);
      float t1 = a0 * (acc[i][2][e] + bvj[2]) + a1 * (acc[i][3][e] + bvj[3]);
#pragma unroll
      for (int x = 1; x < 16; x <<= 1) {
        t0 += __shfl_xor(t0, x, 32);
        t1 += __shfl_xor(t1, x, 32);
      }
      if (lidx == 0) {
        int tok = m0 + row;
        int b = tok >> 10, s = tok & 1023;
        size_t base = (((size_t)b * 16 + h) * 1024 + s) << 6;
        X[base + d0]     = t0 * 0.125f;
        X[base + d0 + 1] = t1 * 0.125f;
      }
    }
}

// ---------------------------------------------------------------------------
// 4) Output projection: out = X @ Wo^T + bo   (X read flat as (2048, 1024))
// ---------------------------------------------------------------------------
__global__ void __launch_bounds__(256) k_oproj(const float* __restrict__ X,
                                               const float* __restrict__ Wo,
                                               const float* __restrict__ bo,
                                               float* __restrict__ out)
{
  DECL_GEMM_LDS
  const int m0 = blockIdx.y * 128;
  const int n0 = blockIdx.x * 128;
  v8f acc[2][4];
  gemm_core_bf16x3(X, Wo, 1024, m0, n0, Ah, Al, Bh, Bl, acc);

  const int lane = threadIdx.x & 31;
  const int w    = threadIdx.x >> 5;
  const int lh   = lane >> 4;
  const int lidx = lane & 15;
  const int rbase = (w & 3) << 5;
  const int cbase = (w >> 2) << 6;
#pragma unroll
  for (int j = 0; j < 4; ++j) {
    int n = n0 + cbase + 16 * j + lidx;
    float bj = bo[n];
#pragma unroll
    for (int i = 0; i < 2; ++i)
#pragma unroll
      for (int e = 0; e < 8; ++e) {
        int row = m0 + rbase + 16 * i + e + 8 * lh;
        out[(size_t)row * 1024 + n] = acc[i][j][e] + bj;
      }
  }
}

// ---------------------------------------------------------------------------
extern "C" void kernel_launch(void* const* d_in, const int* in_sizes, int n_in,
                              void* d_out, int out_size, void* d_ws, size_t ws_size,
                              hipStream_t stream)
{
  const float* query = (const float*)d_in[0];
  // d_in[1] = key (unused by forward)
  const float* value = (const float*)d_in[2];
  const float* rk    = (const float*)d_in[3];
  const float* rw    = (const float*)d_in[4];
  const float* Wq    = (const float*)d_in[5];
  const float* bq    = (const float*)d_in[6];
  const float* Wv    = (const float*)d_in[7];
  const float* bv    = (const float*)d_in[8];
  const float* Wo    = (const float*)d_in[9];
  const float* bo    = (const float*)d_in[10];
  float* out = (float*)d_out;

  float* qp   = (float*)d_ws;            // 2048*1024 fp32 ( 8 MB)
  float* attn = qp + 2048 * 1024;        // 2048*16*32  ( 4 MB)
  float* X    = attn + 2048 * 16 * 32;   // 2048*1024   ( 8 MB)

  dim3 blk(256);
  k_qproj<<<dim3(1024 / 128, 2048 / 128), blk, 0, stream>>>(query, Wq, bq, qp);
  k_fuzzy<<<dim3(2048 * 16 / 8), blk, 0, stream>>>(qp, rk, rw, attn);
  k_vproj<<<dim3(32768 / 128, 2048 / 128), blk, 0, stream>>>(value, Wv, bv, attn, X);
  k_oproj<<<dim3(1024 / 128, 2048 / 128), blk, 0, stream>>>(X, Wo, bo, out);
}
